// SelectiveSSM_17995912970478
// MI455X (gfx1250) — matrix operand order repaired and checked by reading, not compiled, not run on hardware
//
#include <hip/hip_runtime.h>
#include <hip/hip_bf16.h>

#define D_MODEL 1024
#define D_INNER 2048
#define D_STATE 16
#define DT_RANK 64
#define D_CONV  4
#define BATCH   2
#define SEQLEN  2048

typedef __attribute__((ext_vector_type(16))) __bf16 v16bf;
typedef __attribute__((ext_vector_type(8)))  float  v8f;

// ---- fast math helpers (native v_exp_f32 / v_log_f32 are base-2) ----
#define LOG2E 1.44269504088896340736f
#define LN2   0.69314718055994530942f

__device__ __forceinline__ float fast_sigmoid(float v) {
    return __builtin_amdgcn_rcpf(1.f + __builtin_amdgcn_exp2f(-v * LOG2E));
}
__device__ __forceinline__ float siluf(float v) { return v * fast_sigmoid(v); }
__device__ __forceinline__ float softplusf(float v) {
    if (v > 20.f) return v;
    return __builtin_amdgcn_logf(1.f + __builtin_amdgcn_exp2f(v * LOG2E)) * LN2;
}

// =====================================================================
// bf16 WMMA GEMM:  C[M,N] (f32) = A[M,K] (bf16, stride lda) @ W[N,K]^T (bf16)
// One wave -> (16*MT) x (16*NT) tile; K stepped by 32.
// Inner loop: full fragment-load phase (2A + 4B b128 clauses), then the
// 8-WMMA chain -- lets the scheduler overlap step k+1 loads with step k math.
// EPI==1: C = softplus(C + bias[n])
// =====================================================================
template<int MT, int NT, int EPI>
__global__ void wmma_gemm_bf16(const __bf16* __restrict__ A,
                               const __bf16* __restrict__ W,
                               float* __restrict__ C,
                               const float* __restrict__ bias,
                               int M, int N, int K, int lda, int ldc)
{
    const int lane = threadIdx.x;          // 0..31
    const int half = lane >> 4;            // 0 | 1
    const int lm   = lane & 15;
    const int mt0  = (blockIdx.x * blockDim.y + threadIdx.y) * MT;
    const int m0   = mt0 * 16;
    const int n0   = blockIdx.y * (16 * NT);
    if (m0 >= M) return;

    v8f acc[MT][NT] = {};

    union Frag { uint4 q[2]; v16bf v; };

    // A fragment per ISA 16-bit A 16x32 layout:
    //   lanes 0-15 : K = {0..7, 16..23};  lanes 16-31 : K = {8..15, 24..31}
    const __bf16* arow[MT];
#pragma unroll
    for (int mi = 0; mi < MT; ++mi)
        arow[mi] = A + (size_t)(m0 + mi * 16 + lm) * lda + half * 8;

    // B fragment: lane -> column n, K = {0..7, 16..23} (half 0) / {8..15, 24..31} (half 1), same as A.
    // Per-tile base pointers hoisted out of the K loop (address math once).
    const __bf16* wptr[NT];
#pragma unroll
    for (int t = 0; t < NT; ++t) {
        int n = n0 + t * 16 + lm;
        n = (n < N) ? n : (N - 1);         // clamp OOB reads; stores guarded below
        wptr[t] = W + (size_t)n * K + half * 8;
    }

    for (int k0 = 0; k0 < K; k0 += 32) {
        // ---- memory phase: everything this K-step needs ----
        Frag af[MT], bf[NT];
#pragma unroll
        for (int mi = 0; mi < MT; ++mi) {
            af[mi].q[0] = *(const uint4*)(arow[mi] + k0);        // K = kb..kb+7
            af[mi].q[1] = *(const uint4*)(arow[mi] + k0 + 16);   // K = kb+16..+23
        }
#pragma unroll
        for (int t = 0; t < NT; ++t) {
            bf[t].q[0] = *(const uint4*)(wptr[t] + k0);
            bf[t].q[1] = *(const uint4*)(wptr[t] + k0 + 16);
        }
#pragma unroll
        for (int mi = 0; mi < MT; ++mi)
            __builtin_prefetch(arow[mi] + k0 + 64, 0, 1);        // global_prefetch_b8

        // ---- compute phase: MT*NT WMMAs ----
#pragma unroll
        for (int t = 0; t < NT; ++t)
#pragma unroll
            for (int mi = 0; mi < MT; ++mi)
                acc[mi][t] = __builtin_amdgcn_wmma_f32_16x16x32_bf16(
                    false, af[mi].v, false, bf[t].v, (short)0, acc[mi][t], false, false);
    }

    // C/D layout: VGPR r -> row m0 + mi*16 + r + 8*half, col n0 + t*16 + lm
#pragma unroll
    for (int mi = 0; mi < MT; ++mi)
#pragma unroll
        for (int t = 0; t < NT; ++t) {
            int n = n0 + t * 16 + lm;
            if (n >= N) continue;
#pragma unroll
            for (int r = 0; r < 8; ++r) {
                int m = m0 + mi * 16 + r + half * 8;
                float v = acc[mi][t][r];
                if (EPI == 1) v = softplusf(v + bias[n]);
                C[(size_t)m * ldc + n] = v;
            }
        }
}

// ---------------------------------------------------------------------
// One-time f32 -> bf16 operand conversion
// ---------------------------------------------------------------------
__global__ void cvt_bf16_kernel(const float* __restrict__ src,
                                __bf16* __restrict__ dst, int n)
{
    int i = blockIdx.x * blockDim.x + threadIdx.x;
    if (i < n) dst[i] = (__bf16)src[i];
}

// ---------------------------------------------------------------------
// Depthwise causal conv1d (4 taps) + SiLU. x lives in cols [0,2048) of xz.
// Writes f32 (for the scan) and bf16 (for the x_proj GEMM).
// ---------------------------------------------------------------------
__global__ void conv_silu_kernel(const float* __restrict__ xz,
                                 const float* __restrict__ conv_w,
                                 const float* __restrict__ conv_b,
                                 float* __restrict__ xcf,
                                 __bf16* __restrict__ xcb)
{
    size_t tid = (size_t)blockIdx.x * blockDim.x + threadIdx.x;
    if (tid >= (size_t)BATCH * SEQLEN * D_INNER) return;
    int d = (int)(tid & (D_INNER - 1));
    int l = (int)((tid >> 11) & (SEQLEN - 1));
    int b = (int)(tid >> 22);
    float acc = conv_b[d];
#pragma unroll
    for (int k = 0; k < D_CONV; ++k) {
        int li = l + k - (D_CONV - 1);
        if (li >= 0)
            acc += conv_w[d * D_CONV + k] *
                   xz[((size_t)b * SEQLEN + li) * (2 * D_INNER) + d];
    }
    float v = siluf(acc);
    xcf[tid] = v;
    xcb[tid] = (__bf16)v;
}

// ---------------------------------------------------------------------
// Selective scan: lane owns one (b,d) channel, 16 states in registers,
// sequential over L, zero synchronization. B/C broadcast via wave32
// shuffles; fused D-skip and SiLU(z) gate; emits bf16 for out_proj.
// ---------------------------------------------------------------------
__global__ void scan_kernel(const float* __restrict__ xz,     // z in cols [2048,4096)
                            const float* __restrict__ xc,
                            const float* __restrict__ proj,   // (M,96): [dt_lo | B | C]
                            const float* __restrict__ dt,
                            const float* __restrict__ A_log,
                            const float* __restrict__ Dp,
                            __bf16* __restrict__ y)
{
    int tid  = blockIdx.x * blockDim.x + threadIdx.x;  // 0..BATCH*D_INNER-1
    int d    = tid & (D_INNER - 1);
    int b    = tid >> 11;
    int lane = threadIdx.x & 31;

    float A2[D_STATE], h[D_STATE];
#pragma unroll
    for (int n = 0; n < D_STATE; ++n) {
        // A = -exp(A_log); pre-scale by log2(e) so hot loop uses native exp2
        A2[n] = -__builtin_amdgcn_exp2f(A_log[d * D_STATE + n] * LOG2E) * LOG2E;
        h[n]  = 0.f;
    }
    const float Dd = Dp[d];

    for (int l = 0; l < SEQLEN; ++l) {
        size_t row = (size_t)b * SEQLEN + l;
        float xv  = xc[row * D_INNER + d];
        float dtv = dt[row * D_INNER + d];
        // whole wave shares (b,l): lanes cooperatively fetch B[0..15],C[0..15]
        const float* p = proj + row * (DT_RANK + 2 * D_STATE);
        float bcv = (lane < 16) ? p[DT_RANK + lane]
                                : p[DT_RANK + D_STATE + (lane - 16)];
        float s    = dtv * xv;
        float yacc = 0.f;
#pragma unroll
        for (int n = 0; n < D_STATE; ++n) {
            float Bn = __shfl(bcv, n, 32);
            float Cn = __shfl(bcv, n + 16, 32);
            float dA = __builtin_amdgcn_exp2f(dtv * A2[n]);
            h[n] = dA * h[n] + s * Bn;
            yacc += h[n] * Cn;
        }
        float zv = xz[row * (2 * D_INNER) + D_INNER + d];
        y[row * D_INNER + d] = (__bf16)((yacc + Dd * xv) * siluf(zv));
    }
}

// =====================================================================
extern "C" void kernel_launch(void* const* d_in, const int* in_sizes, int n_in,
                              void* d_out, int out_size, void* d_ws, size_t ws_size,
                              hipStream_t stream)
{
    (void)in_sizes; (void)n_in; (void)out_size; (void)ws_size;
    const float* hs     = (const float*)d_in[0];
    const float* W_in   = (const float*)d_in[1];
    const float* conv_w = (const float*)d_in[2];
    const float* conv_b = (const float*)d_in[3];
    const float* W_x    = (const float*)d_in[4];
    const float* W_dt   = (const float*)d_in[5];
    const float* b_dt   = (const float*)d_in[6];
    const float* A_log  = (const float*)d_in[7];
    const float* D_par  = (const float*)d_in[8];
    const float* W_out  = (const float*)d_in[9];
    float* out = (float*)d_out;

    const int M  = BATCH * SEQLEN;           // 4096
    const int NP = DT_RANK + 2 * D_STATE;    // 96

    // ---- workspace layout: f32 region, then bf16 region ----
    float* ws   = (float*)d_ws;
    float* xz   = ws;                         // M * 4096 f32
    float* xcf  = xz   + (size_t)M * 4096;    // M * 2048 f32
    float* proj = xcf  + (size_t)M * 2048;    // M * 96   f32
    float* dtb  = proj + (size_t)M * NP;      // M * 2048 f32

    __bf16* bw      = (__bf16*)(dtb + (size_t)M * 2048);
    __bf16* hs_bf   = bw;                                      // M * 1024
    __bf16* Win_bf  = hs_bf   + (size_t)M * D_MODEL;           // 4096 * 1024
    __bf16* Wx_bf   = Win_bf  + (size_t)2 * D_INNER * D_MODEL; // 96 * 2048
    __bf16* Wdt_bf  = Wx_bf   + (size_t)NP * D_INNER;          // 2048 * 64
    __bf16* Wout_bf = Wdt_bf  + (size_t)D_INNER * DT_RANK;     // 1024 * 2048
    __bf16* xc_bf   = Wout_bf + (size_t)D_MODEL * D_INNER;     // M * 2048
    __bf16* proj_bf = xc_bf   + (size_t)M * D_INNER;           // M * 96
    __bf16* y_bf    = proj_bf + (size_t)M * NP;                // M * 2048

    // ---- 0) one-time operand conversions to bf16 ----
    {
        struct { const float* s; __bf16* d; int n; } cv[5] = {
            { hs,    hs_bf,   M * D_MODEL },
            { W_in,  Win_bf,  2 * D_INNER * D_MODEL },
            { W_x,   Wx_bf,   NP * D_INNER },
            { W_dt,  Wdt_bf,  D_INNER * DT_RANK },
            { W_out, Wout_bf, D_MODEL * D_INNER },
        };
        for (int i = 0; i < 5; ++i)
            cvt_bf16_kernel<<<(cv[i].n + 255) / 256, 256, 0, stream>>>(
                cv[i].s, cv[i].d, cv[i].n);
    }

    dim3 blk(32, 4);   // 4 waves / block, each owns a 32x64 output tile

    // 1) in_proj: xz = hs @ W_in^T           (4096 x 4096 x 1024)
    wmma_gemm_bf16<2, 4, 0><<<dim3(M / 128, (2 * D_INNER) / 64), blk, 0, stream>>>(
        hs_bf, Win_bf, xz, nullptr, M, 2 * D_INNER, D_MODEL, D_MODEL, 2 * D_INNER);

    // 2) depthwise conv + SiLU (f32 + bf16 outputs)
    {
        size_t total = (size_t)M * D_INNER;
        conv_silu_kernel<<<(unsigned)((total + 255) / 256), 256, 0, stream>>>(
            xz, conv_w, conv_b, xcf, xc_bf);
    }

    // 3) x_proj: proj = xc @ W_x^T           (4096 x 96 x 2048)
    wmma_gemm_bf16<2, 2, 0><<<dim3(M / 128, NP / 32), blk, 0, stream>>>(
        xc_bf, Wx_bf, proj, nullptr, M, NP, D_INNER, D_INNER, NP);

    // 3b) proj -> bf16 (operand of the dt GEMM; scan keeps reading f32 proj)
    {
        int n = M * NP;
        cvt_bf16_kernel<<<(n + 255) / 256, 256, 0, stream>>>(proj, proj_bf, n);
    }

    // 4) dt = softplus(proj[:, :64] @ W_dt^T + b_dt)   (4096 x 2048 x 64)
    wmma_gemm_bf16<2, 4, 1><<<dim3(M / 128, D_INNER / 64), blk, 0, stream>>>(
        proj_bf, Wdt_bf, dtb, b_dt, M, D_INNER, DT_RANK, NP, D_INNER);

    // 5) selective scan + skip + gate (emits bf16 y)
    scan_kernel<<<(BATCH * D_INNER) / 256, 256, 0, stream>>>(
        xz, xcf, proj, dtb, A_log, D_par, y_bf);

    // 6) out_proj: out = y @ W_out^T         (4096 x 1024 x 2048)
    wmma_gemm_bf16<2, 4, 0><<<dim3(M / 128, D_MODEL / 64), blk, 0, stream>>>(
        y_bf, Wout_bf, out, nullptr, M, D_MODEL, D_INNER, D_INNER, D_MODEL);
}
